// GraphTP_22067541967732
// MI455X (gfx1250) — matrix operand order repaired and checked by reading, not compiled, not run on hardware
//
#include <hip/hip_runtime.h>
#include <hip/hip_bf16.h>
#include <math.h>

// ---------------------------------------------------------------------------
// CDNA5 (gfx1250) implementation. wave32, WMMA 16x16x32 bf16, f32 accumulate.
// ---------------------------------------------------------------------------

typedef __attribute__((ext_vector_type(16))) __bf16 v16bf;
typedef __attribute__((ext_vector_type(2)))  __bf16 v2bf;
typedef __attribute__((ext_vector_type(2)))  float  v2f;
typedef __attribute__((ext_vector_type(8)))  float  v8f;

#define N_SC   384
#define S_SENT 64
#define DIM_IN 768
#define HL     256
#define SSZ    512
#define T_MOD  40
#define A_INF  128
#define V_INF  2048
#define PADW   400
#define KMAXN  6

// Packed f32x2 -> bf16x2 (lowers to v_cvt_pk_bf16_f32 on gfx1250)
static __device__ __forceinline__ unsigned pk_bf16(float a, float b) {
  v2f x; x[0] = a; x[1] = b;
  union { v2bf v; unsigned u; } r;
  r.v = __builtin_convertvector(x, v2bf);
  return r.u;
}

static __device__ __forceinline__ unsigned short bf16s(float f) {
  union { __bf16 b; unsigned short s; } r;
  r.b = (__bf16)f;
  return r.s;
}

union BF16Frag { v16bf v; unsigned short s[16]; unsigned u[8]; };

static __device__ __forceinline__ v8f wmma_bf16(const BF16Frag& a, const BF16Frag& b, v8f c) {
  return __builtin_amdgcn_wmma_f32_16x16x32_bf16(false, a.v, false, b.v, (short)0, c, false, false);
}

static __device__ __forceinline__ float sigf(float x) { return 1.f / (1.f + expf(-x)); }

// ---------------------------------------------------------------------------
// Generic WMMA GEMM: out[M,Nc] = act( A[M,K] @ Bmat + bias + resid )
//   transW==0: Bmat = W^T with W row-major [Nc,K]   (out = A @ W.T)
//   transW==1: Bmat = W    with W row-major [K,Nc]  (out = A @ W)
// Block = 128 threads (4 waves); tile: 16 rows x 64 cols; K multiple of 32.
// ---------------------------------------------------------------------------
__global__ __launch_bounds__(128)
void wmma_gemm_kernel(const float* __restrict__ A, int lda,
                      const float* __restrict__ W, int ldw, int transW,
                      const float* __restrict__ bias,
                      const float* __restrict__ resid, int ldr,
                      float* __restrict__ out, int ldo,
                      int K, int act) {
  const int wave = threadIdx.x >> 5;
  const int lane = threadIdx.x & 31;
  const int g    = lane >> 4;          // lane half
  const int rsel = lane & 15;
  const int row_base = blockIdx.y << 4;
  const int col_base = (blockIdx.x << 6) + (wave << 4);

  v8f acc = {};
  const float* arow = A + (size_t)(row_base + rsel) * lda;
  for (int k0 = 0; k0 < K; k0 += 32) {
    BF16Frag af, bf;
    // A 16x32 bf16 layout: lane half g: s[i] -> K = k0+16g+i
    const int ka = k0 + (g << 4);
    if (k0 + 32 < K) __builtin_prefetch(arow + k0 + 32, 0, 1);
#pragma unroll
    for (int i = 0; i < 4; ++i) {
      af.u[i]     = pk_bf16(arow[ka + 2 * i],      arow[ka + 2 * i + 1]);
      af.u[4 + i] = pk_bf16(arow[ka + 8 + 2 * i], arow[ka + 8 + 2 * i + 1]);
    }
    // B 32x16 bf16 layout: lane half g: s[i] -> B[k0+16g+i][col]
    const int kb = k0 + (g << 4);
    if (!transW) {
      const float* wrow = W + (size_t)(col_base + rsel) * ldw;
#pragma unroll
      for (int i = 0; i < 8; ++i) bf.u[i] = pk_bf16(wrow[kb + 2 * i], wrow[kb + 2 * i + 1]);
    } else {
      const float* wcol = W + col_base + rsel;
      const int base = kb * ldw;   // fits in int for all trans GEMMs here
#pragma unroll
      for (int i = 0; i < 8; ++i)
        bf.u[i] = pk_bf16(wcol[base + 2 * i * ldw], wcol[base + (2 * i + 1) * ldw]);
    }
    acc = wmma_bf16(af, bf, acc);
  }
  // C/D layout: element (m,n): lane = n + 16*(m/8), vgpr = m%8
  const int n = rsel;
  const int mbase = g << 3;
  const float bval = bias ? bias[col_base + n] : 0.f;
#pragma unroll
  for (int v = 0; v < 8; ++v) {
    const int m = mbase + v;
    float val = acc[v] + bval;
    if (resid) val += resid[(size_t)(row_base + m) * ldr + col_base + n];
    if (act == 1) val = tanhf(val);
    out[(size_t)(row_base + m) * ldo + col_base + n] = val;
  }
}

// ---------------------------------------------------------------------------
// f32 -> bf16 conversion (for pre-converting Whh)
// ---------------------------------------------------------------------------
__global__ void cvt_bf16_kernel(const float* __restrict__ x, unsigned short* __restrict__ y, int count) {
  int i = blockIdx.x * blockDim.x + threadIdx.x;
  if (i < count) y[i] = bf16s(x[i]);
}

// ---------------------------------------------------------------------------
// Scene BiLSTM recurrent scan. grid = (24, 2[dir]), block = 256 (8 waves).
// Each block owns 16 scenes; h kept in LDS as bf16; c kept in accumulator-
// layout VGPRs. Wave w owns hidden cols [32w,32w+32) across all 4 gates, so
// i,f,g,o of a hidden unit live in the same lane -> no cross-wave exchange.
// ---------------------------------------------------------------------------
__global__ __launch_bounds__(256)
void scene_lstm_kernel(const float* __restrict__ XG,             // [N*S, 2048] gates precomp (+bias)
                       const unsigned short* __restrict__ WhhBF, // [2,1024,256] bf16
                       float* __restrict__ script_outs)          // [N*S, 512]
{
  __shared__ __align__(16) unsigned short h_lds[16][256];
  const int dir    = blockIdx.y;
  const int scene0 = blockIdx.x << 4;
  const int wave   = threadIdx.x >> 5;
  const int lane   = threadIdx.x & 31;
  const int g      = lane >> 4;
  const int rsel   = lane & 15;
  const int hc0    = wave << 5;
  const unsigned short* Whh = WhhBF + (size_t)dir * 1024 * 256;

  for (int i = threadIdx.x; i < 16 * 256; i += 256) (&h_lds[0][0])[i] = 0;
  v8f cst[2];
#pragma unroll
  for (int j = 0; j < 2; ++j)
#pragma unroll
    for (int v = 0; v < 8; ++v) cst[j][v] = 0.f;
  __syncthreads();

  for (int t = 0; t < S_SENT; ++t) {
    const int tt = dir ? (S_SENT - 1 - t) : t;
    v8f gacc[4][2];
    const unsigned* hu = (const unsigned*)&h_lds[rsel][0];
#pragma unroll
    for (int gi = 0; gi < 4; ++gi) {
#pragma unroll
      for (int j = 0; j < 2; ++j) {
        v8f acc = {};
        const int col = gi * 256 + hc0 + (j << 4) + rsel;     // Whh row == gate col
        const unsigned* wu = (const unsigned*)(Whh + (size_t)col * 256);
        for (int kt = 0; kt < 8; ++kt) {
          BF16Frag af, bf;
          const int ha = kt * 16 + (g << 3);
#pragma unroll
          for (int i = 0; i < 4; ++i) { af.u[i] = hu[ha + i]; af.u[4 + i] = hu[ha + 4 + i]; }
          const int wb = kt * 16 + (g << 3);
#pragma unroll
          for (int i = 0; i < 8; ++i) bf.u[i] = wu[wb + i];
          acc = wmma_bf16(af, bf, acc);
        }
        gacc[gi][j] = acc;
      }
    }
    __syncthreads();   // all h_lds reads complete
    const int mbase = g << 3;
#pragma unroll
    for (int j = 0; j < 2; ++j) {
      const int hcol = hc0 + (j << 4) + rsel;
#pragma unroll
      for (int v = 0; v < 8; ++v) {
        const int m = mbase + v;
        const size_t xrow = (size_t)(scene0 + m) * S_SENT + tt;
        const float* xg = XG + xrow * 2048 + (size_t)dir * 1024;
        const float gi_ = gacc[0][j][v] + xg[0 * 256 + hcol];
        const float gf_ = gacc[1][j][v] + xg[1 * 256 + hcol];
        const float gg_ = gacc[2][j][v] + xg[2 * 256 + hcol];
        const float go_ = gacc[3][j][v] + xg[3 * 256 + hcol];
        const float c = sigf(gf_) * cst[j][v] + sigf(gi_) * tanhf(gg_);
        const float h = sigf(go_) * tanhf(c);
        cst[j][v] = c;
        h_lds[m][hcol] = bf16s(h);
        script_outs[xrow * 512 + (size_t)dir * 256 + hcol] = h;
      }
    }
    __syncthreads();   // new h visible
  }
}

// ---------------------------------------------------------------------------
// Script LSTM (batch=1, T=384). grid = 2 (dir), block = 256. Latency-bound.
// ---------------------------------------------------------------------------
__global__ __launch_bounds__(256)
void script_lstm_kernel(const float* __restrict__ XGs,   // [384,2048]
                        const float* __restrict__ WhhAll,// [2,1024,256] f32
                        float* __restrict__ script_emb)  // [384,512]
{
  __shared__ float h[256];
  __shared__ float c[256];
  const int dir = blockIdx.x;
  const int j = threadIdx.x;
  const float* Whh = WhhAll + (size_t)dir * 1024 * 256;
  h[j] = 0.f; c[j] = 0.f;
  __syncthreads();
  for (int t = 0; t < N_SC; ++t) {
    const int tt = dir ? (N_SC - 1 - t) : t;
    const float* xg = XGs + (size_t)tt * 2048 + (size_t)dir * 1024;
    float gate[4];
#pragma unroll
    for (int gi = 0; gi < 4; ++gi) {
      const float* w = Whh + (size_t)(gi * 256 + j) * 256;
      float acc = xg[gi * 256 + j];
      for (int k = 0; k < 256; ++k) acc += h[k] * w[k];
      gate[gi] = acc;
    }
    __syncthreads();
    const float cv = sigf(gate[1]) * c[j] + sigf(gate[0]) * tanhf(gate[2]);
    const float hv = sigf(gate[3]) * tanhf(cv);
    c[j] = cv; h[j] = hv;
    script_emb[(size_t)tt * 512 + (size_t)dir * 256 + j] = hv;
    __syncthreads();
  }
}

// ---------------------------------------------------------------------------
// Sentence attention pooling -> scene_emb. grid=384, block=256.
// ---------------------------------------------------------------------------
__global__ __launch_bounds__(256)
void attnpool_scene_kernel(const float* __restrict__ U,       // [N*S,512] tanh'ed
                           const float* __restrict__ attn_v,
                           const float* __restrict__ script_outs,
                           const int* __restrict__ scene_lens,
                           float* __restrict__ scene_emb) {
  __shared__ float red[256];
  __shared__ float ev[64];
  __shared__ float sa[64];
  const int n = blockIdx.x, t = threadIdx.x;
  {
    const int s = t >> 2, part = t & 3;
    const float* u = U + ((size_t)n * S_SENT + s) * 512 + part * 128;
    float acc = 0.f;
    for (int i = 0; i < 128; ++i) acc += u[i] * attn_v[part * 128 + i];
    red[t] = acc;
  }
  __syncthreads();
  if ((t & 3) == 0) ev[t >> 2] = red[t] + red[t + 1] + red[t + 2] + red[t + 3];
  __syncthreads();
  int len = scene_lens[n]; if (len < 1) len = 1;
  if (t < 64) {
    float m = -3.0e38f;
    for (int s = 0; s < len; ++s) m = fmaxf(m, ev[s]);
    float sum = 0.f;
    for (int s = 0; s < len; ++s) sum += expf(ev[s] - m);
    sa[t] = (t < len) ? expf(ev[t] - m) / sum : 0.f;
  }
  __syncthreads();
  for (int hh = t; hh < 512; hh += 256) {
    float acc = 0.f;
    for (int s = 0; s < S_SENT; ++s)
      acc += sa[s] * script_outs[((size_t)n * S_SENT + s) * 512 + hh];
    scene_emb[(size_t)n * 512 + hh] = acc;
  }
}

// ---------------------------------------------------------------------------
// Modality attention pooling (+ optional L2 normalize). grid=384, block=256.
// ---------------------------------------------------------------------------
__global__ __launch_bounds__(256)
void modpool_kernel(const float* __restrict__ X, const float* __restrict__ w,
                    const float* __restrict__ b, int T,
                    float* __restrict__ outv, int normalize) {
  __shared__ float red[256];
  __shared__ float ev[64];
  __shared__ float sa[64];
  const int n = blockIdx.x, t = threadIdx.x;
  {
    const int s = t >> 2, part = t & 3;
    float acc = 0.f;
    if (s < T) {
      const float* x = X + ((size_t)n * T + s) * 512 + part * 128;
      for (int i = 0; i < 128; ++i) acc += x[i] * w[part * 128 + i];
    }
    red[t] = acc;
  }
  __syncthreads();
  if ((t & 3) == 0 && (t >> 2) < T)
    ev[t >> 2] = tanhf(red[t] + red[t + 1] + red[t + 2] + red[t + 3] + b[0]);
  __syncthreads();
  if (t < T) {
    float m = -3.0e38f;
    for (int s = 0; s < T; ++s) m = fmaxf(m, ev[s]);
    float sum = 0.f;
    for (int s = 0; s < T; ++s) sum += expf(ev[s] - m);
    sa[t] = expf(ev[t] - m) / sum;
  }
  __syncthreads();
  float o0 = 0.f, o1 = 0.f;
  for (int s = 0; s < T; ++s) {
    const float* x = X + ((size_t)n * T + s) * 512;
    o0 += sa[s] * x[t];
    o1 += sa[s] * x[t + 256];
  }
  if (normalize) {
    red[t] = o0 * o0 + o1 * o1;
    __syncthreads();
    for (int off = 128; off > 0; off >>= 1) { if (t < off) red[t] += red[t + off]; __syncthreads(); }
    const float nrm = fmaxf(sqrtf(red[0]), 1e-12f);
    o0 /= nrm; o1 /= nrm;
  }
  outv[(size_t)n * 512 + t] = o0;
  outv[(size_t)n * 512 + t + 256] = o1;
}

// ---------------------------------------------------------------------------
// Softmax over 384 elements of row r of strided matrix. out row-major [384,384].
// ---------------------------------------------------------------------------
__global__ __launch_bounds__(256)
void softmax384_kernel(const float* __restrict__ X, int s1, int s2, float* __restrict__ out) {
  __shared__ float red[256];
  const int r = blockIdx.x, t = threadIdx.x;
  const float a = X[(size_t)r * s1 + (size_t)t * s2];
  const float bv = (t < 128) ? X[(size_t)r * s1 + (size_t)(t + 256) * s2] : -3.0e38f;
  red[t] = fmaxf(a, bv);
  __syncthreads();
  for (int off = 128; off > 0; off >>= 1) { if (t < off) red[t] = fmaxf(red[t], red[t + off]); __syncthreads(); }
  const float m = red[0];
  __syncthreads();
  const float ea = expf(a - m);
  const float eb = (t < 128) ? expf(bv - m) : 0.f;
  red[t] = ea + eb;
  __syncthreads();
  for (int off = 128; off > 0; off >>= 1) { if (t < off) red[t] += red[t + off]; __syncthreads(); }
  const float s = red[0];
  out[(size_t)r * 384 + t] = ea / s;
  if (t < 128) out[(size_t)r * 384 + t + 256] = eb / s;
}

// ---------------------------------------------------------------------------
// core/nei scalar heads. grid=384, block=256.
// ---------------------------------------------------------------------------
__global__ __launch_bounds__(256)
void corenei_kernel(const float* __restrict__ scene_emb,
                    const float* __restrict__ core_w, const float* __restrict__ core_b,
                    const float* __restrict__ nei_w,  const float* __restrict__ nei_b,
                    float* __restrict__ core, float* __restrict__ neiv) {
  __shared__ float r1[256];
  __shared__ float r2[256];
  const int n = blockIdx.x, t = threadIdx.x;
  const float* x = scene_emb + (size_t)n * 512;
  r1[t] = x[t] * core_w[t] + x[t + 256] * core_w[t + 256];
  r2[t] = x[t] * nei_w[t]  + x[t + 256] * nei_w[t + 256];
  __syncthreads();
  for (int off = 128; off > 0; off >>= 1) {
    if (t < off) { r1[t] += r1[t + off]; r2[t] += r2[t + off]; }
    __syncthreads();
  }
  if (t == 0) { core[n] = tanhf(r1[0] + core_b[0]); neiv[n] = tanhf(r2[0] + nei_b[0]); }
}

__global__ __launch_bounds__(256)
void colmean_kernel(const float* __restrict__ mm, const float* __restrict__ core,
                    const float* __restrict__ neiv, float* __restrict__ colmean) {
  __shared__ float red[256];
  const int jc = blockIdx.x, t = threadIdx.x;
  float acc = core[t] * mm[(size_t)t * 384 + jc];
  if (t < 128) acc += core[t + 256] * mm[(size_t)(t + 256) * 384 + jc];
  red[t] = acc;
  __syncthreads();
  for (int off = 128; off > 0; off >>= 1) { if (t < off) red[t] += red[t + off]; __syncthreads(); }
  if (t == 0) colmean[jc] = neiv[jc] * red[0] / 384.f;
}

__global__ void scores_kernel(const float* __restrict__ mm, const float* __restrict__ core,
                              const float* __restrict__ neiv, const float* __restrict__ colmean,
                              float* __restrict__ scores) {
  const int idx = blockIdx.x * blockDim.x + threadIdx.x;
  if (idx >= N_SC * PADW) return;
  const int i = idx / PADW, jc = idx % PADW;
  scores[idx] = (jc < N_SC) ? core[i] * neiv[jc] * mm[(size_t)i * 384 + jc] - colmean[jc] : 0.f;
}

__global__ void neidec_kernel(const float* __restrict__ scores, const float* __restrict__ Wd,
                              const float* __restrict__ bd, float* __restrict__ logits) {
  const int idx = blockIdx.x * blockDim.x + threadIdx.x;
  if (idx >= N_SC * KMAXN) return;
  const int n = idx / KMAXN, k = idx % KMAXN;
  const float* srow = scores + (size_t)n * PADW;
  const float* wrow = Wd + (size_t)k * PADW;
  float acc = bd[k];
  for (int p = 0; p < PADW; ++p) acc += srow[p] * wrow[p];
  logits[idx] = acc;
}

__global__ void topk_kernel(const float* __restrict__ logits, const float* __restrict__ scores,
                            int* __restrict__ ksel, int* __restrict__ idxout) {
  const int n = blockIdx.x * blockDim.x + threadIdx.x;
  if (n >= N_SC) return;
  const float* lg = logits + (size_t)n * KMAXN;
  int best = 0; float bv = lg[0];
  for (int k = 1; k < KMAXN; ++k) if (lg[k] > bv) { bv = lg[k]; best = k; }
  ksel[n] = best + 1;
  const float* srow = scores + (size_t)n * PADW;
  int chosen[KMAXN];
  for (int p = 0; p < KMAXN; ++p) {
    float bm = -3.4e38f; int bj = 0;
    for (int jc = 0; jc < N_SC; ++jc) {
      bool used = false;
      for (int q = 0; q < p; ++q) if (chosen[q] == jc) used = true;
      if (used) continue;
      const float v = srow[jc];
      if (v > bm) { bm = v; bj = jc; }
    }
    chosen[p] = bj;
    idxout[(size_t)n * KMAXN + p] = bj;
  }
}

__global__ __launch_bounds__(512)
void gather_kernel(const float* __restrict__ xw, const int* __restrict__ idx,
                   const int* __restrict__ ksel, const float* __restrict__ gcn_b,
                   float* __restrict__ graph_out) {
  const int n = blockIdx.x, hh = threadIdx.x;
  float acc = gcn_b[hh];
  const int k = ksel[n];
  for (int q = 0; q < KMAXN; ++q)
    if (q < k) acc += xw[(size_t)idx[(size_t)n * KMAXN + q] * 512 + hh];
  graph_out[(size_t)n * 512 + hh] = acc;
}

__global__ void tp_kernel(const float* __restrict__ script_emb, const float* __restrict__ graph_out,
                          const float* __restrict__ tpW, const float* __restrict__ tpb,
                          float* __restrict__ logits) {
  const int idx = blockIdx.x * blockDim.x + threadIdx.x;
  if (idx >= N_SC * 5) return;
  const int n = idx / 5, cc = idx % 5;
  const float* w = tpW + (size_t)cc * 1024;
  const float* a = script_emb + (size_t)n * 512;
  const float* gr = graph_out + (size_t)n * 512;
  float acc = tpb[cc];
  for (int hh = 0; hh < 512; ++hh) acc += a[hh] * w[hh] + gr[hh] * w[512 + hh];
  logits[idx] = acc;
}

__global__ __launch_bounds__(256)
void final_softmax_kernel(const float* __restrict__ logits, float* __restrict__ out) {
  __shared__ float red[256];
  const int cc = blockIdx.x, t = threadIdx.x;
  const float a = logits[(size_t)t * 5 + cc];
  const float b = (t < 128) ? logits[(size_t)(t + 256) * 5 + cc] : -3.0e38f;
  red[t] = fmaxf(a, b);
  __syncthreads();
  for (int off = 128; off > 0; off >>= 1) { if (t < off) red[t] = fmaxf(red[t], red[t + off]); __syncthreads(); }
  const float m = red[0];
  __syncthreads();
  const float ea = expf(a - m);
  const float eb = (t < 128) ? expf(b - m) : 0.f;
  red[t] = ea + eb;
  __syncthreads();
  for (int off = 128; off > 0; off >>= 1) { if (t < off) red[t] += red[t + off]; __syncthreads(); }
  const float s = red[0];
  out[(size_t)cc * 384 + t] = ea / s;
  if (t < 128) out[(size_t)cc * 384 + t + 256] = eb / s;
}

// ---------------------------------------------------------------------------
// Host side
// ---------------------------------------------------------------------------
static inline void gemm(hipStream_t st, const float* A, int lda, const float* W, int ldw, int tw,
                        const float* bias, const float* resid, int ldr,
                        float* out, int ldo, int M, int K, int Nc, int act) {
  dim3 g(Nc / 64, M / 16), b(128);
  wmma_gemm_kernel<<<g, b, 0, st>>>(A, lda, W, ldw, tw, bias, resid, ldr, out, ldo, K, act);
}

extern "C" void kernel_launch(void* const* d_in, const int* in_sizes, int n_in,
                              void* d_out, int out_size, void* d_ws, size_t ws_size,
                              hipStream_t stream) {
  const float* script = (const float*)d_in[0];
  const float* audio  = (const float*)d_in[1];
  const float* vision = (const float*)d_in[2];
  const int*   lens   = (const int*)d_in[3];
  const float* scWih  = (const float*)d_in[4];   // [2,1024,768]
  const float* scWhh  = (const float*)d_in[5];   // [2,1024,256]
  const float* scB    = (const float*)d_in[6];   // [2,1024]
  const float* attnW  = (const float*)d_in[7];
  const float* attnB  = (const float*)d_in[8];
  const float* attnV  = (const float*)d_in[9];
  const float* spWih  = (const float*)d_in[10];
  const float* spWhh  = (const float*)d_in[11];
  const float* spB    = (const float*)d_in[12];
  const float* apW    = (const float*)d_in[13];
  const float* apB    = (const float*)d_in[14];
  const float* vpW    = (const float*)d_in[15];
  const float* vpB    = (const float*)d_in[16];
  const float* aaW    = (const float*)d_in[17];
  const float* aaB    = (const float*)d_in[18];
  const float* avW    = (const float*)d_in[19];
  const float* avB    = (const float*)d_in[20];
  const float* coreW  = (const float*)d_in[21];
  const float* coreB  = (const float*)d_in[22];
  const float* neiW   = (const float*)d_in[23];
  const float* neiB   = (const float*)d_in[24];
  const float* ndW    = (const float*)d_in[25];
  const float* ndB    = (const float*)d_in[26];
  const float* gcnW   = (const float*)d_in[27];
  const float* gcnB   = (const float*)d_in[28];
  const float* tpW    = (const float*)d_in[29];
  const float* tpB    = (const float*)d_in[30];
  (void)in_sizes; (void)n_in; (void)out_size; (void)ws_size;

  char* ws = (char*)d_ws;
  // Big regions (with reuse):
  float* XG          = (float*)(ws + 0);                 // [24576,2048]  201.3MB
  float* script_outs = (float*)(ws + 201326592);         // [24576,512]   50.3MB
  float* U           = (float*)(ws + 251658240);         // [24576,512]   50.3MB
  float* XGs         = U;                                // reuse after attn pool
  float* aud         = (float*)(ws + 0);                 // reuse XG region after scene LSTM
  float* vis         = (float*)(ws + 31457280);
  size_t off = 301989888;
  float* scene_emb  = (float*)(ws + off); off += 786432;
  float* script_emb = (float*)(ws + off); off += 786432;
  float* aud_v      = (float*)(ws + off); off += 786432;
  float* vis_v      = (float*)(ws + off); off += 786432;
  float* va         = (float*)(ws + off); off += 589824;
  float* P1         = (float*)(ws + off); off += 589824;
  float* P2         = (float*)(ws + off); off += 589824;
  float* avbuf      = (float*)(ws + off); off += 1572864;   // [384,1024] concat(aud_att,vis_att)
  float* mmb        = (float*)(ws + off); off += 589824;
  float* coreb      = (float*)(ws + off); off += 1536;
  float* neib       = (float*)(ws + off); off += 1536;
  float* cmean      = (float*)(ws + off); off += 1536;
  float* scores     = (float*)(ws + off); off += 614400;    // [384,400]
  float* nlog       = (float*)(ws + off); off += 9216;
  int*   ksel       = (int*)(ws + off);   off += 1536;
  int*   tidx       = (int*)(ws + off);   off += 9216;
  float* xw         = (float*)(ws + off); off += 786432;
  float* graph_out  = (float*)(ws + off); off += 786432;
  float* tplog      = (float*)(ws + off); off += 7680;
  unsigned short* whhBF = (unsigned short*)(ws + off); off += 1048576;  // [2,1024,256] bf16

  // 0) pre-convert scene Whh to bf16
  cvt_bf16_kernel<<<2048, 256, 0, stream>>>(scWhh, whhBF, 2 * 1024 * 256);
  // 1) scene gate pre-activations: XG = script @ Wih(2dirs stacked).T + b
  gemm(stream, script, DIM_IN, scWih, DIM_IN, 0, scB, nullptr, 0, XG, 2048, N_SC * S_SENT, DIM_IN, 2048, 0);
  // 2) scene BiLSTM recurrent scan (WMMA on h@Whh.T)
  scene_lstm_kernel<<<dim3(N_SC / 16, 2), 256, 0, stream>>>(XG, whhBF, script_outs);
  // 3) attention pre-activation U = tanh(script_outs @ attn_W.T + attn_b)
  gemm(stream, script_outs, SSZ, attnW, SSZ, 0, attnB, nullptr, 0, U, SSZ, N_SC * S_SENT, SSZ, SSZ, 1);
  // 4) masked softmax pooling -> scene_emb
  attnpool_scene_kernel<<<N_SC, 256, 0, stream>>>(U, attnV, script_outs, lens, scene_emb);
  // 5) script gate pre-activations
  gemm(stream, scene_emb, SSZ, spWih, SSZ, 0, spB, nullptr, 0, XGs, 2048, N_SC, SSZ, 2048, 0);
  // 6) script BiLSTM (batch 1, latency path)
  script_lstm_kernel<<<2, 256, 0, stream>>>(XGs, spWhh, script_emb);
  // 7) modality projections (tanh)
  gemm(stream, audio, A_INF, apW, A_INF, 0, apB, nullptr, 0, aud, SSZ, N_SC * T_MOD, A_INF, SSZ, 1);
  gemm(stream, vision, V_INF, vpW, V_INF, 0, vpB, nullptr, 0, vis, SSZ, N_SC * T_MOD, V_INF, SSZ, 1);
  // 8) modality attention pooling + l2 normalize
  modpool_kernel<<<N_SC, 256, 0, stream>>>(aud, aaW, aaB, T_MOD, aud_v, 1);
  modpool_kernel<<<N_SC, 256, 0, stream>>>(vis, avW, avB, T_MOD, vis_v, 1);
  // 9) cross-modal attention
  gemm(stream, vis_v, SSZ, aud_v, SSZ, 0, nullptr, nullptr, 0, va, N_SC, N_SC, SSZ, N_SC, 0);
  softmax384_kernel<<<N_SC, 256, 0, stream>>>(va, 384, 1, P1);   // row softmax
  softmax384_kernel<<<N_SC, 256, 0, stream>>>(va, 1, 384, P2);   // col softmax (transposed)
  // aud_att = P2 @ vis_v + aud_v -> av[:, :512] ; vis_att = P1 @ aud_v + vis_v -> av[:, 512:]
  gemm(stream, P2, N_SC, vis_v, SSZ, 1, nullptr, aud_v, SSZ, avbuf, 1024, N_SC, N_SC, SSZ, 0);
  gemm(stream, P1, N_SC, aud_v, SSZ, 1, nullptr, vis_v, SSZ, avbuf + 512, 1024, N_SC, N_SC, SSZ, 0);
  // mm = av @ av.T
  gemm(stream, avbuf, 1024, avbuf, 1024, 0, nullptr, nullptr, 0, mmb, N_SC, N_SC, 1024, N_SC, 0);
  // 10) neighbor scoring
  corenei_kernel<<<N_SC, 256, 0, stream>>>(scene_emb, coreW, coreB, neiW, neiB, coreb, neib);
  colmean_kernel<<<N_SC, 256, 0, stream>>>(mmb, coreb, neib, cmean);
  scores_kernel<<<(N_SC * PADW + 255) / 256, 256, 0, stream>>>(mmb, coreb, neib, cmean, scores);
  neidec_kernel<<<(N_SC * KMAXN + 255) / 256, 256, 0, stream>>>(scores, ndW, ndB, nlog);
  topk_kernel<<<(N_SC + 63) / 64, 64, 0, stream>>>(nlog, scores, ksel, tidx);
  // 11) GCN: xw = script_emb @ gcn_W ; gather + bias
  gemm(stream, script_emb, SSZ, gcnW, SSZ, 1, nullptr, nullptr, 0, xw, SSZ, N_SC, SSZ, SSZ, 0);
  gather_kernel<<<N_SC, 512, 0, stream>>>(xw, tidx, ksel, gcnB, graph_out);
  // 12) TP heads + softmax over scenes
  tp_kernel<<<(N_SC * 5 + 255) / 256, 256, 0, stream>>>(script_emb, graph_out, tpW, tpB, tplog);
  final_softmax_kernel<<<5, 256, 0, stream>>>(tplog, (float*)d_out);
}
